// Encoder_16466904613474
// MI455X (gfx1250) — compile-verified, matrix operand order checked by reading, not run on hardware
//
#include <hip/hip_runtime.h>
#include <hip/hip_bf16.h>

// CDNA5 / gfx1250 fused top-k SAE forward.
// encode GEMM via v_wmma_f32_16x16x32_bf16 with bf16x3 error compensation,
// fused per-row top-16 selection, fused sparse f32 decode.

typedef __attribute__((ext_vector_type(16))) __bf16 v16bf;
typedef __attribute__((ext_vector_type(8)))  __bf16 v8bf;
typedef __attribute__((ext_vector_type(8)))  float  v8f;

#define D_IN    2048
#define D_HID   16384
#define TOPK    16
#define BATCH   4096
#define ROWS_WG 16            // batch rows per workgroup
#define CHUNK   128           // hidden columns per chunk
#define NCHUNK  (D_HID / CHUNK)
#define THREADS 256           // 8 wave32s
#define NEG_INF (-__builtin_inff())

// ---------------- prep: split W (f32) -> bf16 hi + bf16 lo in workspace ----
__global__ void split_w_kernel(const float* __restrict__ W,
                               __bf16* __restrict__ Whi,
                               __bf16* __restrict__ Wlo) {
  size_t e = ((size_t)blockIdx.x * blockDim.x + threadIdx.x) * 8;
  v8bf h, l;
#pragma unroll
  for (int q = 0; q < 8; ++q) {
    float v = W[e + q];
    __bf16 hh = (__bf16)v;
    h[q] = hh;
    l[q] = (__bf16)(v - (float)hh);
  }
  *(v8bf*)(Whi + e) = h;
  *(v8bf*)(Wlo + e) = l;
}

// ---------------- main fused kernel ---------------------------------------
// LDS layout (141312 bytes):
//   [0,      65536)  x_hi  bf16 [16][2048]   } overlaid after GEMM by
//   [65536, 131072)  x_lo  bf16 [16][2048]   } cand_v/cand_i [16][256]
//   [131072,139264)  scores f32 [16][128]
//   [139264,140288)  topv  f32 [16][16]
//   [140288,141312)  topi  i32 [16][16]
__launch_bounds__(THREADS)
__global__ void sae_topk_kernel(const float* __restrict__ x,
                                const float* __restrict__ W,     // f32 [D_HID][D_IN]
                                const float* __restrict__ bias,  // [D_HID]
                                const __bf16* __restrict__ Whi,
                                const __bf16* __restrict__ Wlo,
                                float* __restrict__ out,         // [BATCH][D_IN]
                                int* __restrict__ idx_out) {     // [BATCH][TOPK]
  __shared__ __attribute__((aligned(16))) unsigned char smem[141312];
  __bf16* xhi    = (__bf16*)smem;
  __bf16* xlo    = (__bf16*)(smem + 65536);
  float*  scores = (float*)(smem + 131072);
  float*  topv   = (float*)(smem + 139264);
  int*    topi   = (int*)(smem + 140288);
  float*  cand_v = (float*)smem;               // overlays xhi after GEMM
  int*    cand_i = (int*)(smem + 16384);       // overlays xhi after GEMM

  const int t    = threadIdx.x;
  const int wave = t >> 5;
  const int lane = t & 31;
  const int b0   = blockIdx.x * ROWS_WG;

  // ---- load x tile and split into bf16 hi/lo in LDS ----
#pragma unroll
  for (int i = 0; i < (ROWS_WG * D_IN) / THREADS; ++i) {   // 128 elems/thread
    int e = i * THREADS + t;
    int r = e >> 11, c = e & (D_IN - 1);
    float v = x[(size_t)(b0 + r) * D_IN + c];
    __bf16 h = (__bf16)v;
    xhi[e] = h;
    xlo[e] = (__bf16)(v - (float)h);
  }
  __syncthreads();

  // ---- private running top-16 (per scan thread) ----
  float pv[TOPK]; int pi[TOPK];
#pragma unroll
  for (int q = 0; q < TOPK; ++q) { pv[q] = NEG_INF; pi[q] = 0; }
  float pmin = NEG_INF; int pslot = 0;

  const int ln15  = lane & 15;
  const int khalf = (lane < 16) ? 0 : 8;     // 16-bit A/B operand layout
  const int mbase = (lane < 16) ? 0 : 8;     // 32-bit C/D layout

  for (int chunk = 0; chunk < NCHUNK; ++chunk) {
    const int ncol = chunk * CHUNK + wave * 16 + ln15;   // this lane's B column
    const __bf16* bh = Whi + (size_t)ncol * D_IN;
    const __bf16* bl = Wlo + (size_t)ncol * D_IN;

    v8f acc = {};
#pragma unroll 2
    for (int kk = 0; kk < D_IN; kk += 32) {
      const int ka = kk + khalf;
      // A fragments: lane holds row M=ln15, K groups {ka..ka+7},{ka+16..ka+23}
      v8bf a0 = *(const v8bf*)&xhi[ln15 * D_IN + ka];
      v8bf a1 = *(const v8bf*)&xhi[ln15 * D_IN + ka + 16];
      v8bf l0 = *(const v8bf*)&xlo[ln15 * D_IN + ka];
      v8bf l1 = *(const v8bf*)&xlo[ln15 * D_IN + ka + 16];
      // B fragments: lane holds col N=ncol, same K groups (row-major W)
      v8bf b0v = *(const v8bf*)(bh + ka);
      v8bf b1v = *(const v8bf*)(bh + ka + 16);
      v8bf c0v = *(const v8bf*)(bl + ka);
      v8bf c1v = *(const v8bf*)(bl + ka + 16);
      v16bf Ah, Al, Bh, Bl;
#pragma unroll
      for (int q = 0; q < 8; ++q) {
        Ah[q] = a0[q];  Ah[q + 8] = a1[q];
        Al[q] = l0[q];  Al[q + 8] = l1[q];
        Bh[q] = b0v[q]; Bh[q + 8] = b1v[q];
        Bl[q] = c0v[q]; Bl[q + 8] = c1v[q];
      }
      // bf16x3: (xh+xl)*(Wh+Wl) ~= xh*Wh + xh*Wl + xl*Wh  (f32 accumulate)
      acc = __builtin_amdgcn_wmma_f32_16x16x32_bf16(false, Ah, false, Bh,
                                                    (short)0, acc, false, false);
      acc = __builtin_amdgcn_wmma_f32_16x16x32_bf16(false, Ah, false, Bl,
                                                    (short)0, acc, false, false);
      acc = __builtin_amdgcn_wmma_f32_16x16x32_bf16(false, Al, false, Bh,
                                                    (short)0, acc, false, false);
    }

    // ---- epilogue: scores (+bias) to LDS staging ----
    const float bv  = bias[ncol];
    const int colL  = wave * 16 + ln15;
#pragma unroll
    for (int q = 0; q < 8; ++q)
      scores[(mbase + q) * CHUNK + colL] = acc[q] + bv;
    __syncthreads();

    // ---- distributed top-k scan: 16 threads/row, 8 values each ----
    {
      const int row = t >> 4, sub = t & 15;
#pragma unroll
      for (int j = 0; j < 8; ++j) {
        const int cl = sub * 8 + j;
        float v = scores[row * CHUNK + cl];
        if (v > pmin) {
          pv[pslot] = v; pi[pslot] = chunk * CHUNK + cl;
          pmin = pv[0]; pslot = 0;
#pragma unroll
          for (int q = 1; q < TOPK; ++q)
            if (pv[q] < pmin) { pmin = pv[q]; pslot = q; }
        }
      }
    }
    __syncthreads();   // scores consumed before next chunk's epilogue
  }

  // ---- dump per-thread candidates to LDS (overlays x tile; GEMM is done) --
  {
    const int row = t >> 4, sub = t & 15;
#pragma unroll
    for (int q = 0; q < TOPK; ++q) {
      cand_v[row * 256 + sub * 16 + q] = pv[q];
      cand_i[row * 256 + sub * 16 + q] = pi[q];
    }
  }
  __syncthreads();

  // ---- final 256 -> 16 selection per row (descending, like lax.top_k) ----
  if (t < ROWS_WG) {
    const int row = t;
    for (int k = 0; k < TOPK; ++k) {
      float best = NEG_INF; int bi = 0, bp = 0;
      for (int q = 0; q < 256; ++q) {
        float v = cand_v[row * 256 + q];
        if (v > best) { best = v; bi = cand_i[row * 256 + q]; bp = q; }
      }
      cand_v[row * 256 + bp] = NEG_INF;
      topv[row * TOPK + k] = best;
      topi[row * TOPK + k] = bi;
    }
  }
  __syncthreads();

  // ---- sparse decode in f32: out[r] = sum_k val * W[idx,:] ----
#pragma unroll 2
  for (int i = 0; i < (ROWS_WG * D_IN) / THREADS; ++i) {   // 128 elems/thread
    int e = i * THREADS + t;
    int r = e >> 11, c = e & (D_IN - 1);
    float a = 0.f;
#pragma unroll
    for (int k = 0; k < TOPK; ++k) {
      float v = topv[r * TOPK + k];
      int   g = topi[r * TOPK + k];
      a += v * W[(size_t)g * D_IN + c];
    }
    out[(size_t)(b0 + r) * D_IN + c] = a;
  }

  // ---- idx output: 256 threads == 16 rows x 16 k ----
  {
    const int r = t >> 4, k = t & 15;
    idx_out[(b0 + r) * TOPK + k] = topi[r * TOPK + k];
  }
}

// ---------------- host entry ----------------------------------------------
extern "C" void kernel_launch(void* const* d_in, const int* in_sizes, int n_in,
                              void* d_out, int out_size, void* d_ws, size_t ws_size,
                              hipStream_t stream) {
  const float* x    = (const float*)d_in[0];   // [BATCH][D_IN]
  const float* Wenc = (const float*)d_in[1];   // [D_HID][D_IN]
  const float* benc = (const float*)d_in[2];   // [D_HID]
  // d_in[3] (w_emb_w) == Wenc^T; we gather rows of Wenc instead (contiguous).

  __bf16* Whi = (__bf16*)d_ws;                        // 64 MB
  __bf16* Wlo = Whi + (size_t)D_HID * D_IN;           // 64 MB

  float* out     = (float*)d_out;                               // [BATCH][D_IN]
  int*   idx_out = (int*)(out + (size_t)BATCH * D_IN);          // [BATCH][TOPK]

  const int split_blocks = (D_HID * D_IN) / (8 * 256);          // 8 elems/thread
  split_w_kernel<<<split_blocks, 256, 0, stream>>>(Wenc, Whi, Wlo);
  sae_topk_kernel<<<BATCH / ROWS_WG, THREADS, 0, stream>>>(
      x, Wenc, benc, Whi, Wlo, out, idx_out);
}
